// SubEncoder_23347442221505
// MI455X (gfx1250) — matrix operand order repaired and checked by reading, not compiled, not run on hardware
//
#include <hip/hip_runtime.h>

typedef __attribute__((ext_vector_type(16))) _Float16 v16h;
typedef __attribute__((ext_vector_type(8)))  float    v8f;
typedef __attribute__((ext_vector_type(8)))  _Float16 h8;
typedef __attribute__((ext_vector_type(2)))  _Float16 h2;
typedef __attribute__((ext_vector_type(4)))  float    f4;

#define TOK    16384     // B*N
#define DMODEL 256
#define EDIM   4096
#define NHEAD  8
#define NLAYER 4
#define NNODE  64
#define DHEAD  32

// -------------------------------------------------------------------------
// CDNA5 async global->LDS copy (ASYNCcnt-tracked DMA, no VGPR round trip).
// LDS dest address is taken from a VGPR; generic-pointer low 32 bits == LDS offset.
// -------------------------------------------------------------------------
__device__ __forceinline__ void async_copy_b128(void* lds_dst, const void* gsrc)
{
  unsigned lo = (unsigned)(unsigned long long)lds_dst;
  asm volatile("global_load_async_to_lds_b128 %0, %1, off"
               :: "v"(lo), "v"(gsrc) : "memory");
}
__device__ __forceinline__ void wait_async0()
{
  asm volatile("s_wait_asynccnt 0x0" ::: "memory");
}

// -------------------------------------------------------------------------
// Embedding sum: H[row,d] = vidx_emb[vidx] + pos_emb[pos] + deg_emb[deg]
// -------------------------------------------------------------------------
__global__ __launch_bounds__(256) void embed_kernel(
    const int* __restrict__ vidx, const int* __restrict__ pos, const int* __restrict__ deg,
    const float* __restrict__ ve, const float* __restrict__ pe, const float* __restrict__ de,
    float* __restrict__ H)
{
  int row = blockIdx.x, d = threadIdx.x;
  H[(size_t)row*DMODEL + d] =
      ve[(size_t)vidx[row]*DMODEL + d] +
      pe[(size_t)pos[row]*DMODEL + d] +
      de[(size_t)deg[row]*DMODEL + d];
}

// -------------------------------------------------------------------------
// Pack wq/wk/wv [L,H,D,DH] -> f16 [L][D][768], col = which*256 + h*32 + k
// -------------------------------------------------------------------------
__global__ __launch_bounds__(256) void pack_qkv_kernel(
    const float* __restrict__ wq, const float* __restrict__ wk, const float* __restrict__ wv,
    _Float16* __restrict__ out)
{
  int col = blockIdx.x % 768;
  int l   = blockIdx.x / 768;
  int d   = threadIdx.x;
  int which = col >> 8;
  int rem   = col & 255;
  int h = rem >> 5, kk = rem & 31;
  const float* src = (which == 0) ? wq : (which == 1) ? wk : wv;
  float v = src[(((size_t)l*NHEAD + h)*DMODEL + d)*DHEAD + kk];
  out[((size_t)l*DMODEL + d)*768 + col] = (_Float16)v;
}

__global__ __launch_bounds__(256) void cvt_kernel(
    const float* __restrict__ in, _Float16* __restrict__ out, int n)
{
  int i = blockIdx.x*256 + threadIdx.x;
  if (i < n) out[i] = (_Float16)in[i];
}

// -------------------------------------------------------------------------
// Row LayerNorm, one block (256 thr) per row; epilogue selected by MODE:
//  0: f16 out          1: relu -> f16 out       2: tanh + mask fuse -> f32 out
//  3: f32 + f16 out    4: f16 out               5: f32 out
// -------------------------------------------------------------------------
template<int MODE>
__global__ __launch_bounds__(256) void ln_rows_kernel(
    const float* __restrict__ x, const float* __restrict__ g, const float* __restrict__ bb,
    float* __restrict__ outf, _Float16* __restrict__ outh,
    const int* __restrict__ mask, int rowlen)
{
  __shared__ float s1[256], s2[256];
  int row = blockIdx.x, tid = threadIdx.x;
  const float* xr = x + (size_t)row * rowlen;
  float a = 0.f, q = 0.f;
  for (int i = tid; i < rowlen; i += 256) { float v = xr[i]; a += v; q += v*v; }
  s1[tid] = a; s2[tid] = q; __syncthreads();
  for (int st = 128; st > 0; st >>= 1) {
    if (tid < st) { s1[tid] += s1[tid+st]; s2[tid] += s2[tid+st]; }
    __syncthreads();
  }
  float mean = s1[0] / rowlen;
  float var  = s2[0] / rowlen - mean*mean;
  float rstd = rsqrtf(var + 1e-5f);
  for (int i = tid; i < rowlen; i += 256) {
    float v = (xr[i] - mean) * rstd * g[i] + bb[i];
    size_t o = (size_t)row * rowlen + i;
    if      (MODE == 0) outh[o] = (_Float16)v;
    else if (MODE == 1) outh[o] = (_Float16)fmaxf(v, 0.f);
    else if (MODE == 2) { v = tanhf(v); if (mask[o] == 0) v = -1e10f; outf[o] = v; }
    else if (MODE == 3) { outf[o] = v; outh[o] = (_Float16)v; }
    else if (MODE == 4) outh[o] = (_Float16)v;
    else                outf[o] = v;
  }
}

// -------------------------------------------------------------------------
// Tiled WMMA GEMM: C[M,N] = A[M,K](f16) x B[K,N](TB) (+bias)(+resid)(relu)
// BM=128 BN=128 BK=32, 256 thr = 8 waves, wave tile 32x64 (2x4 WMMA/iter).
// Double-buffered LDS; A tile prefetched with async global->LDS DMA while
// WMMAs consume the current tile; B tile staged transposed (f32 weights are
// converted to f16 in VALU during staging -- DMA cannot convert/transpose).
// -------------------------------------------------------------------------
template<typename TB>
__global__ __launch_bounds__(256) void gemm_kernel(
    const _Float16* __restrict__ A, const TB* __restrict__ Bm,
    const float* __restrict__ bias, const float* __restrict__ resid,
    float* __restrict__ outf, _Float16* __restrict__ outh,
    int M, int Nn, int K, int relu)
{
  const int LDT = 48;                       // f16 stride, 96B rows (16B aligned)
  __shared__ _Float16 As[2][128 * 48];      // 2 x 12 KB
  __shared__ _Float16 Bs[2][128 * 48];      // 2 x 12 KB
  int tid = threadIdx.x;
  int m0 = blockIdx.y * 128, n0 = blockIdx.x * 128;
  int w = tid >> 5, lane = tid & 31;
  int wr = (w & 3) * 32, wc = (w >> 2) * 64;
  int half = lane >> 4, l15 = lane & 15;
  v8f acc[2][4] = {};

  auto stageA = [&](int bf, int k0) {       // 128x32 f16 via async DMA
    #pragma unroll
    for (int it = 0; it < 2; ++it) {
      int v = tid + it * 256;               // 0..511 x 16B
      int r = v >> 2, c = (v & 3) * 8;
      async_copy_b128(&As[bf][r * LDT + c], A + (size_t)(m0 + r) * K + k0 + c);
    }
  };
  auto stageB = [&](int bf, int k0) {       // 32x128 -> transposed Bs[n][k]
    if (sizeof(TB) == 2) {
      #pragma unroll
      for (int it = 0; it < 2; ++it) {
        int v = tid + it * 256;             // 0..511 vectors of 8 f16
        int k = v >> 4, nb = (v & 15) * 8;
        h8 bv = *(const h8*)((const _Float16*)Bm + (size_t)(k0 + k) * Nn + n0 + nb);
        #pragma unroll
        for (int j = 0; j < 8; ++j) Bs[bf][(nb + j) * LDT + k] = bv[j];
      }
    } else {
      #pragma unroll
      for (int it = 0; it < 4; ++it) {
        int v = tid + it * 256;             // 0..1023 vectors of 4 f32
        int k = v >> 5, nb = (v & 31) * 4;
        f4 bv = *(const f4*)((const float*)Bm + (size_t)(k0 + k) * Nn + n0 + nb);
        #pragma unroll
        for (int j = 0; j < 4; ++j) Bs[bf][(nb + j) * LDT + k] = (_Float16)((float)bv[j]);
      }
    }
  };

  stageA(0, 0); stageB(0, 0);
  wait_async0();
  __syncthreads();

  int buf = 0;
  for (int k0 = 0; k0 < K; k0 += 32) {
    if (k0 + 32 < K) { stageA(buf ^ 1, k0 + 32); stageB(buf ^ 1, k0 + 32); }

    v16h af[2], bfv[4];
    #pragma unroll
    for (int t = 0; t < 2; ++t) {
      const _Float16* ap = &As[buf][(wr + t * 16 + l15) * LDT];
      #pragma unroll
      for (int i = 0; i < 8; ++i) {
        int kb = 2 * (i & 3) + 16 * (i >> 2) + 8 * half;  // ISA A 16x32 f16 layout
        h2 p = *(const h2*)&ap[kb];
        af[t][2*i] = p[0]; af[t][2*i + 1] = p[1];
      }
    }
    #pragma unroll
    for (int c = 0; c < 4; ++c) {
      const _Float16* bp = &Bs[buf][(wc + c * 16 + l15) * LDT];
      #pragma unroll
      for (int i = 0; i < 8; ++i) {
        int kb = 2 * (i & 3) + 16 * (i >> 2) + 8 * half;  // same k order as A frag
        h2 p = *(const h2*)&bp[kb];
        bfv[c][2*i] = p[0]; bfv[c][2*i + 1] = p[1];
      }
    }
    #pragma unroll
    for (int r = 0; r < 2; ++r)
      #pragma unroll
      for (int c = 0; c < 4; ++c)
        acc[r][c] = __builtin_amdgcn_wmma_f32_16x16x32_f16(
            false, af[r], false, bfv[c], (short)0, acc[r][c], false, false);

    wait_async0();        // prefetch for next tile complete (overlapped with WMMA)
    __syncthreads();
    buf ^= 1;
  }

  // ---- epilogue (C layout: M = vi + 8*half, N = l15) ----
  #pragma unroll
  for (int r = 0; r < 2; ++r)
    #pragma unroll
    for (int c = 0; c < 4; ++c)
      #pragma unroll
      for (int vi = 0; vi < 8; ++vi) {
        int gm = m0 + wr + r * 16 + vi + 8 * half;
        int gn = n0 + wc + c * 16 + l15;
        float val = acc[r][c][vi];
        if (bias)  val += bias[gn];
        if (resid) val += resid[(size_t)gm * Nn + gn];
        if (relu)  val = fmaxf(val, 0.f);
        size_t o = (size_t)gm * Nn + gn;
        if (outf) outf[o] = val;
        if (outh) outh[o] = (_Float16)val;
      }
}

// -------------------------------------------------------------------------
// Fused attention: 1 block per batch elem, 1 wave per head (N=64, DH=32).
// bias/mask tile DMA'd to LDS once (async), shared by all 8 heads.
// S=QK^T (16 WMMA) -> softmax in regs (shfl within 16-lane half) -> P via
// per-head LDS scratch -> O=PV (16 WMMA).
// -------------------------------------------------------------------------
__global__ __launch_bounds__(256) void attn_kernel(
    const _Float16* __restrict__ QKV, const float* __restrict__ biasm,
    _Float16* __restrict__ Oh)
{
  extern __shared__ char smem[];
  float*    bm   = (float*)smem;                       // 64*64 f32 = 16KB
  _Float16* Sall = (_Float16*)(smem + 64 * 64 * 4);    // 8 * 64*72 f16 = 72KB
  int b = blockIdx.x, tid = threadIdx.x;
  int h = tid >> 5, lane = tid & 31;
  int half = lane >> 4, l15 = lane & 15;

  // async DMA bias/mask tile into LDS (16KB = 1024 x 16B)
  #pragma unroll
  for (int it = 0; it < 4; ++it) {
    int v = tid + it * 256;
    async_copy_b128(&bm[v * 4], biasm + (size_t)b * 4096 + v * 4);
  }
  wait_async0();
  __syncthreads();

  _Float16* Sh = Sall + h * 64 * 72;
  const _Float16* base = QKV + (size_t)b * 64 * 768;

  // ---- S = Q K^T ----
  v16h qf[4];
  #pragma unroll
  for (int mi = 0; mi < 4; ++mi) {
    const _Float16* qp = base + (size_t)(mi * 16 + l15) * 768 + h * DHEAD;
    #pragma unroll
    for (int i = 0; i < 8; ++i) {
      int kb = 2 * (i & 3) + 16 * (i >> 2) + 8 * half;
      h2 p = *(const h2*)&qp[kb];
      qf[mi][2*i] = p[0]; qf[mi][2*i + 1] = p[1];
    }
  }
  v8f sacc[4][4] = {};
  #pragma unroll
  for (int nj = 0; nj < 4; ++nj) {
    v16h kf;
    const _Float16* kp = base + (size_t)(nj * 16 + l15) * 768 + 256 + h * DHEAD;
    #pragma unroll
    for (int i = 0; i < 8; ++i) {
      int kb = 2 * (i & 3) + 16 * (i >> 2) + 8 * half;  // B frag of K^T, same k order as Q frag
      h2 p = *(const h2*)&kp[kb];
      kf[2*i] = p[0]; kf[2*i + 1] = p[1];
    }
    #pragma unroll
    for (int mi = 0; mi < 4; ++mi)
      sacc[mi][nj] = __builtin_amdgcn_wmma_f32_16x16x32_f16(
          false, qf[mi], false, kf, (short)0, sacc[mi][nj], false, false);
  }

  // ---- scale + (bias|mask), softmax ----
  const float scale = 0.17677669529663687f;  // 1/sqrt(32)
  #pragma unroll
  for (int mi = 0; mi < 4; ++mi)
    #pragma unroll
    for (int nj = 0; nj < 4; ++nj)
      #pragma unroll
      for (int vi = 0; vi < 8; ++vi) {
        int gm = mi * 16 + vi + 8 * half, gn = nj * 16 + l15;
        sacc[mi][nj][vi] = sacc[mi][nj][vi] * scale + bm[gm * 64 + gn];
      }
  #pragma unroll
  for (int mi = 0; mi < 4; ++mi)
    #pragma unroll
    for (int vi = 0; vi < 8; ++vi) {
      float mx = -3.4e38f;
      #pragma unroll
      for (int nj = 0; nj < 4; ++nj) mx = fmaxf(mx, sacc[mi][nj][vi]);
      #pragma unroll
      for (int m = 1; m < 16; m <<= 1) mx = fmaxf(mx, __shfl_xor(mx, m, 32));
      float sm = 0.f;
      #pragma unroll
      for (int nj = 0; nj < 4; ++nj) {
        float e = __expf(sacc[mi][nj][vi] - mx);
        sacc[mi][nj][vi] = e; sm += e;
      }
      #pragma unroll
      for (int m = 1; m < 16; m <<= 1) sm += __shfl_xor(sm, m, 32);
      float inv = 1.f / sm;
      int row = mi * 16 + vi + 8 * half;
      #pragma unroll
      for (int nj = 0; nj < 4; ++nj)
        Sh[row * 72 + nj * 16 + l15] = (_Float16)(sacc[mi][nj][vi] * inv);
    }

  // ---- O = P V ----
  v8f oacc[4][2] = {};
  #pragma unroll
  for (int kt = 0; kt < 2; ++kt) {
    v16h pf[4];
    #pragma unroll
    for (int mi = 0; mi < 4; ++mi) {
      const _Float16* pp = &Sh[(mi * 16 + l15) * 72 + kt * 32];
      #pragma unroll
      for (int i = 0; i < 8; ++i) {
        int kb = 2 * (i & 3) + 16 * (i >> 2) + 8 * half;
        h2 p = *(const h2*)&pp[kb];
        pf[mi][2*i] = p[0]; pf[mi][2*i + 1] = p[1];
      }
    }
    #pragma unroll
    for (int nt = 0; nt < 2; ++nt) {
      v16h vf;
      const _Float16* vp = base + 512 + h * DHEAD + nt * 16 + l15;
      #pragma unroll
      for (int i = 0; i < 8; ++i) {
        int k = kt * 32 + 2 * (i & 3) + 16 * (i >> 2) + 8 * half;
        vf[2*i]     = vp[(size_t)k * 768];
        vf[2*i + 1] = vp[(size_t)(k + 1) * 768];
      }
      #pragma unroll
      for (int mi = 0; mi < 4; ++mi)
        oacc[mi][nt] = __builtin_amdgcn_wmma_f32_16x16x32_f16(
            false, pf[mi], false, vf, (short)0, oacc[mi][nt], false, false);
    }
  }
  #pragma unroll
  for (int mi = 0; mi < 4; ++mi)
    #pragma unroll
    for (int nt = 0; nt < 2; ++nt)
      #pragma unroll
      for (int vi = 0; vi < 8; ++vi) {
        int row = mi * 16 + vi + 8 * half;
        int col = h * DHEAD + nt * 16 + l15;
        Oh[((size_t)b * 64 + row) * DMODEL + col] = (_Float16)oacc[mi][nt][vi];
      }
}

// -------------------------------------------------------------------------
extern "C" void kernel_launch(void* const* d_in, const int* in_sizes, int n_in,
                              void* d_out, int out_size, void* d_ws, size_t ws_size,
                              hipStream_t stream)
{
  (void)in_sizes; (void)n_in; (void)out_size; (void)ws_size;
  const int*   vidx     = (const int*)  d_in[0];
  const int*   pos      = (const int*)  d_in[1];
  const int*   deg      = (const int*)  d_in[2];
  const float* dist     = (const float*)d_in[3];
  const int*   mask     = (const int*)  d_in[4];
  const float* vidx_emb = (const float*)d_in[5];
  const float* pos_emb  = (const float*)d_in[6];
  const float* deg_emb  = (const float*)d_in[7];
  const float* e_ln1_g  = (const float*)d_in[8];
  const float* e_ln1_b  = (const float*)d_in[9];
  const float* e_w1     = (const float*)d_in[10];
  const float* e_b1     = (const float*)d_in[11];
  const float* e_ln2_g  = (const float*)d_in[12];
  const float* e_ln2_b  = (const float*)d_in[13];
  const float* e_w2     = (const float*)d_in[14];
  const float* e_b2     = (const float*)d_in[15];
  const float* e_ln3_g  = (const float*)d_in[16];
  const float* e_ln3_b  = (const float*)d_in[17];
  const float* gt_ln1_g = (const float*)d_in[18];
  const float* gt_ln1_b = (const float*)d_in[19];
  const float* wq       = (const float*)d_in[20];
  const float* wk       = (const float*)d_in[21];
  const float* wv       = (const float*)d_in[22];
  const float* fc_w     = (const float*)d_in[23];
  const float* gt_ln2_g = (const float*)d_in[24];
  const float* gt_ln2_b = (const float*)d_in[25];
  const float* ffn_w1   = (const float*)d_in[26];
  const float* ffn_b1   = (const float*)d_in[27];
  const float* ffn_w2   = (const float*)d_in[28];
  const float* ffn_b2   = (const float*)d_in[29];
  const float* ln_g     = (const float*)d_in[30];
  const float* ln_b     = (const float*)d_in[31];

  char* wsp = (char*)d_ws;
  auto take = [&](size_t bytes) -> char* {
    char* p = wsp; wsp += (bytes + 255) & ~(size_t)255; return p;
  };
  float*    H     = (float*)   take((size_t)TOK * DMODEL * 4);   // 16 MB
  float*    Hn    = (float*)   take((size_t)TOK * DMODEL * 4);   // 16 MB
  _Float16* Hnh   = (_Float16*)take((size_t)TOK * DMODEL * 2);   //  8 MB
  _Float16* X0h   = (_Float16*)take((size_t)256 * EDIM * 2);     //  2 MB
  float*    Yf    = (float*)   take((size_t)256 * EDIM * 4);     //  4 MB
  _Float16* X1h   = (_Float16*)take((size_t)256 * EDIM * 2);     //  2 MB
  float*    BiasM = (float*)   take((size_t)256 * EDIM * 4);     //  4 MB
  _Float16* QKVh  = (_Float16*)take((size_t)TOK * 768 * 2);      // 24 MB
  _Float16* Oh    = (_Float16*)take((size_t)TOK * DMODEL * 2);   //  8 MB
  _Float16* H2h   = (_Float16*)take((size_t)TOK * DMODEL * 2);   //  8 MB
  _Float16* F1h   = (_Float16*)take((size_t)TOK * DMODEL * 2);   //  8 MB
  _Float16* Wqkvh = (_Float16*)take((size_t)NLAYER * DMODEL * 768 * 2);
  _Float16* Fch   = (_Float16*)take((size_t)NLAYER * DMODEL * DMODEL * 2);
  _Float16* W1h   = (_Float16*)take((size_t)NLAYER * DMODEL * DMODEL * 2);
  _Float16* W2h   = (_Float16*)take((size_t)NLAYER * DMODEL * DMODEL * 2);

  // ---- node features + weight packing ----
  embed_kernel<<<TOK, 256, 0, stream>>>(vidx, pos, deg, vidx_emb, pos_emb, deg_emb, H);
  pack_qkv_kernel<<<NLAYER * 768, 256, 0, stream>>>(wq, wk, wv, Wqkvh);
  const int WSZ = NLAYER * DMODEL * DMODEL;  // 262144
  cvt_kernel<<<WSZ / 256, 256, 0, stream>>>(fc_w,   Fch, WSZ);
  cvt_kernel<<<WSZ / 256, 256, 0, stream>>>(ffn_w1, W1h, WSZ);
  cvt_kernel<<<WSZ / 256, 256, 0, stream>>>(ffn_w2, W2h, WSZ);

  // ---- edge-bias MLP: LN -> GEMM -> LN+ReLU -> GEMM -> LN+tanh+mask ----
  ln_rows_kernel<0><<<256, 256, 0, stream>>>(dist, e_ln1_g, e_ln1_b, nullptr, X0h, nullptr, EDIM);
  gemm_kernel<float><<<dim3(EDIM / 128, 256 / 128), 256, 0, stream>>>(
      X0h, e_w1, e_b1, nullptr, Yf, nullptr, 256, EDIM, EDIM, 0);
  ln_rows_kernel<1><<<256, 256, 0, stream>>>(Yf, e_ln2_g, e_ln2_b, nullptr, X1h, nullptr, EDIM);
  gemm_kernel<float><<<dim3(EDIM / 128, 256 / 128), 256, 0, stream>>>(
      X1h, e_w2, e_b2, nullptr, Yf, nullptr, 256, EDIM, EDIM, 0);
  ln_rows_kernel<2><<<256, 256, 0, stream>>>(Yf, e_ln3_g, e_ln3_b, BiasM, nullptr, mask, EDIM);

  // ---- transformer layers ----
  const size_t attn_smem = 64 * 64 * 4 + (size_t)NHEAD * 64 * 72 * 2;  // 90112 B
  for (int l = 0; l < NLAYER; ++l) {
    ln_rows_kernel<3><<<TOK, 256, 0, stream>>>(
        H, gt_ln1_g + l * DMODEL, gt_ln1_b + l * DMODEL, Hn, Hnh, nullptr, DMODEL);
    gemm_kernel<_Float16><<<dim3(768 / 128, TOK / 128), 256, 0, stream>>>(
        Hnh, Wqkvh + (size_t)l * DMODEL * 768, nullptr, nullptr, nullptr, QKVh,
        TOK, 768, DMODEL, 0);
    attn_kernel<<<256, 256, attn_smem, stream>>>(QKVh, BiasM, Oh);
    gemm_kernel<_Float16><<<dim3(DMODEL / 128, TOK / 128), 256, 0, stream>>>(
        Oh, Fch + (size_t)l * DMODEL * DMODEL, nullptr, Hn, H, nullptr,
        TOK, DMODEL, DMODEL, 0);                         // H = Hn + O @ fc
    ln_rows_kernel<4><<<TOK, 256, 0, stream>>>(
        H, gt_ln2_g + l * DMODEL, gt_ln2_b + l * DMODEL, nullptr, H2h, nullptr, DMODEL);
    gemm_kernel<_Float16><<<dim3(DMODEL / 128, TOK / 128), 256, 0, stream>>>(
        H2h, W1h + (size_t)l * DMODEL * DMODEL, ffn_b1 + l * DMODEL, nullptr, nullptr, F1h,
        TOK, DMODEL, DMODEL, 1);                         // relu(H2@W1+b1) -> f16
    gemm_kernel<_Float16><<<dim3(DMODEL / 128, TOK / 128), 256, 0, stream>>>(
        F1h, W2h + (size_t)l * DMODEL * DMODEL, ffn_b2 + l * DMODEL, H, H, nullptr,
        TOK, DMODEL, DMODEL, 0);                         // H = ... + H
  }

  // ---- final LayerNorm -> d_out (f32) ----
  ln_rows_kernel<5><<<TOK, 256, 0, stream>>>(H, ln_g, ln_b, (float*)d_out, nullptr, nullptr, DMODEL);
}
